// WindowAttention_87943750353536
// MI455X (gfx1250) — matrix operand order repaired and checked by reading, not compiled, not run on hardware
//
#include <hip/hip_runtime.h>
#include <hip/hip_bf16.h>

#define DEVFN __device__ __forceinline__

typedef __attribute__((ext_vector_type(16))) __bf16       v16bf;
typedef __attribute__((ext_vector_type(8)))  float        v8f;
typedef __attribute__((ext_vector_type(8)))  unsigned int v8u;
typedef __attribute__((ext_vector_type(4)))  unsigned int u32x4;

// ---------------- problem sizes ----------------
constexpr int BW   = 2048;       // windows
constexpr int NTOK = 49;         // tokens per window (7x7)
constexpr int C    = 512;
constexpr int H    = 16;
constexpr int HD   = 32;         // head dim
constexpr int C3   = 3 * C;      // 1536
constexpr int MROWS = BW * NTOK; // 100352 (divisible by 128)
constexpr int NP   = 64;         // padded token count for WMMA tiling
constexpr int LSTR = 40;         // padded LDS row stride (bf16) -> bank-conflict-free

// ---------------- workspace layout (bytes) ----------------
constexpr size_t OFF_WQKV  = 0;                                  // bf16 [C3][C]  (transposed: [col][k])
constexpr size_t OFF_WPROJ = OFF_WQKV  + (size_t)C3 * C * 2;     // bf16 [C][C]   (transposed)
constexpr size_t OFF_COS   = OFF_WPROJ + (size_t)C  * C * 2;     // f32  [49][16]
constexpr size_t OFF_SIN   = OFF_COS + 4096;
constexpr size_t QKV_ELEMS = (size_t)BW * H * NTOK * HD;         // 51,380,224
constexpr size_t OFF_Q     = OFF_SIN + 4096;                     // bf16 [BW][H][49][32]
constexpr size_t OFF_K     = OFF_Q + QKV_ELEMS * 2;
constexpr size_t OFF_V     = OFF_K + QKV_ELEMS * 2;
constexpr size_t OFF_AO    = OFF_V + QKV_ELEMS * 2;              // bf16 [MROWS][C]

// ---------------- WMMA fragment helpers ----------------
// A-matrix 16x32 bf16 (ISA 7.12.2): lane-half h: elements 0..7 hold K=8h+0..7,
// elements 8..15 hold K=16+8h+0..7  -> two contiguous 16B runs.
// B-matrix 32x16 bf16: lane = column; same K order as A (K=8h+0..7, 16+8h+0..7).

DEVFN v16bf cat8(u32x4 a, u32x4 b) {
  v8u u = {a[0], a[1], a[2], a[3], b[0], b[1], b[2], b[3]};
  return __builtin_bit_cast(v16bf, u);
}

DEVFN v16bf frag_a_bf16(const __bf16* rowptr, int h) {   // rowptr at K-offset 0
  u32x4 q0 = *reinterpret_cast<const u32x4*>(rowptr + 8 * h);
  u32x4 q1 = *reinterpret_cast<const u32x4*>(rowptr + 16 + 8 * h);
  return cat8(q0, q1);
}

DEVFN v16bf frag_b_bf16(const __bf16* rowptr, int h) {
  u32x4 q0 = *reinterpret_cast<const u32x4*>(rowptr + 8 * h);
  u32x4 q1 = *reinterpret_cast<const u32x4*>(rowptr + 16 + 8 * h);
  return cat8(q0, q1);
}

DEVFN u32x4 pack_bf16x8(float4 a, float4 b) {
  v16bf t;
  t[0] = (__bf16)a.x; t[1] = (__bf16)a.y; t[2] = (__bf16)a.z; t[3] = (__bf16)a.w;
  t[4] = (__bf16)b.x; t[5] = (__bf16)b.y; t[6] = (__bf16)b.z; t[7] = (__bf16)b.w;
  t[8] = t[9] = t[10] = t[11] = t[12] = t[13] = t[14] = t[15] = (__bf16)0.f;
  v8u u = __builtin_bit_cast(v8u, t);
  u32x4 r = {u[0], u[1], u[2], u[3]};
  return r;
}

DEVFN v8f wmma_bf16(v16bf a, v16bf b, v8f c) {
  return __builtin_amdgcn_wmma_f32_16x16x32_bf16(false, a, false, b, (short)0, c,
                                                 false, false);
}

DEVFN v8f vzero8() {
  v8f z = {0.f, 0.f, 0.f, 0.f, 0.f, 0.f, 0.f, 0.f};
  return z;
}

// ---------------- kernel 1: weight transpose->bf16 + RoPE tables ----------------
__global__ void __launch_bounds__(256) wa_prep_kernel(const float* __restrict__ Wqkv,
                                                      const float* __restrict__ Wproj,
                                                      char* __restrict__ ws) {
  __bf16* wq = (__bf16*)(ws + OFF_WQKV);
  __bf16* wp = (__bf16*)(ws + OFF_WPROJ);
  float*  ct = (float*)(ws + OFF_COS);
  float*  st = (float*)(ws + OFF_SIN);
  int idx = blockIdx.x * blockDim.x + threadIdx.x;
  if (idx < C3 * C) {            // wq[c][k] = Wqkv[k][c]
    int c = idx >> 9, k = idx & 511;
    wq[idx] = (__bf16)Wqkv[(size_t)k * C3 + c];
  }
  if (idx < C * C) {             // wp[c][k] = Wproj[k][c]
    int c = idx >> 9, k = idx & 511;
    wp[idx] = (__bf16)Wproj[(size_t)k * C + c];
  }
  if (idx < NTOK * 16) {         // 2D RoPE over 7x7 grid, half=16, quarter=8
    int n = idx >> 4, j = idx & 15;
    int row = n / 7, col = n % 7;
    int jj = j & 7;
    float inv = __powf(10000.f, -(float)(2 * jj) / 16.f);
    float f = (j < 8 ? (float)row : (float)col) * inv;
    ct[idx] = __cosf(f);
    st[idx] = __sinf(f);
  }
}

// ---------------- kernel 2: QKV GEMM (bf16 WMMA, LDS double-buffered) ----------
// Block 128 threads (4 waves) computes a 128x64 tile; wave owns 32x64 (8 WMMA/step).
__global__ void __launch_bounds__(128) wa_qkv_kernel(const float* __restrict__ x,
                                                     const float* __restrict__ bqkv,
                                                     char* __restrict__ ws) {
  const __bf16* wq = (const __bf16*)(ws + OFF_WQKV);
  __bf16* Q = (__bf16*)(ws + OFF_Q);
  __bf16* K = (__bf16*)(ws + OFF_K);
  __bf16* V = (__bf16*)(ws + OFF_V);

  __shared__ __bf16 sA[2][128 * LSTR];  // [row][k0..31] per stage
  __shared__ __bf16 sB[2][64 * LSTR];   // [col][k0..31] per stage

  const int tid  = threadIdx.x;
  const int lane = tid & 31;
  const int wave = tid >> 5;
  const int h    = lane >> 4;
  const int rowblock = blockIdx.y * 128;
  const int colbase  = blockIdx.x * 64;

  // cooperative staging: each thread owns one A row (32 f32) and half a B col (16 bf16)
  const float*  agp = x + (size_t)(rowblock + tid) * C;
  const __bf16* bgp = wq + (size_t)(colbase + (tid >> 1)) * C + (tid & 1) * 16;
  __bf16* sa_row = nullptr;  // unused
  const int sa_off = tid * LSTR;
  const int sb_off = (tid >> 1) * LSTR + (tid & 1) * 16;

  // prologue: stage step 0
  float4 af[8];
  u32x4  bf2[2];
#pragma unroll
  for (int i = 0; i < 8; ++i) af[i] = *reinterpret_cast<const float4*>(agp + 4 * i);
#pragma unroll
  for (int j = 0; j < 2; ++j) bf2[j] = *reinterpret_cast<const u32x4*>(bgp + 8 * j);
#pragma unroll
  for (int i = 0; i < 4; ++i)
    *reinterpret_cast<u32x4*>(&sA[0][sa_off + 8 * i]) = pack_bf16x8(af[2 * i], af[2 * i + 1]);
#pragma unroll
  for (int j = 0; j < 2; ++j)
    *reinterpret_cast<u32x4*>(&sB[0][sb_off + 8 * j]) = bf2[j];
  __syncthreads();

  v8f acc[2][4];
#pragma unroll
  for (int a = 0; a < 2; ++a)
#pragma unroll
    for (int t = 0; t < 4; ++t) acc[a][t] = vzero8();

  for (int s = 0; s < 16; ++s) {
    const int cur = s & 1, nxt = cur ^ 1;
    if (s < 15) {  // issue next step's global loads early (no wait until store phase)
      const int kk = (s + 1) * 32;
#pragma unroll
      for (int i = 0; i < 8; ++i) af[i] = *reinterpret_cast<const float4*>(agp + kk + 4 * i);
#pragma unroll
      for (int j = 0; j < 2; ++j) bf2[j] = *reinterpret_cast<const u32x4*>(bgp + kk + 8 * j);
    }
    // compute current step entirely from LDS
    const __bf16* sa = sA[cur];
    const __bf16* sb = sB[cur];
    v16bf a0 = frag_a_bf16(sa + (wave * 32 + (lane & 15)) * LSTR, h);
    v16bf a1 = frag_a_bf16(sa + (wave * 32 + 16 + (lane & 15)) * LSTR, h);
#pragma unroll
    for (int t = 0; t < 4; ++t) {
      v16bf bb = frag_b_bf16(sb + (t * 16 + (lane & 15)) * LSTR, h);
      acc[0][t] = wmma_bf16(a0, bb, acc[0][t]);
      acc[1][t] = wmma_bf16(a1, bb, acc[1][t]);
    }
    if (s < 15) {  // convert + store next stage
#pragma unroll
      for (int i = 0; i < 4; ++i)
        *reinterpret_cast<u32x4*>(&sA[nxt][sa_off + 8 * i]) = pack_bf16x8(af[2 * i], af[2 * i + 1]);
#pragma unroll
      for (int j = 0; j < 2; ++j)
        *reinterpret_cast<u32x4*>(&sB[nxt][sb_off + 8 * j]) = bf2[j];
    }
    __syncthreads();
  }
  (void)sa_row;

  // epilogue: bias + head-major scatter to Q/K/V
#pragma unroll
  for (int t = 0; t < 4; ++t) {
    int c = colbase + t * 16 + (lane & 15);
    float bias = bqkv[c];
    int three = c >> 9;          // 0=q 1=k 2=v
    int hh    = (c >> 5) & 15;   // head
    int d     = c & 31;
    __bf16* dst = (three == 0) ? Q : ((three == 1) ? K : V);
#pragma unroll
    for (int a = 0; a < 2; ++a) {
#pragma unroll
      for (int r = 0; r < 8; ++r) {
        int grow = rowblock + wave * 32 + a * 16 + r + 8 * h;  // global token row
        int b    = grow / NTOK;
        int n    = grow - b * NTOK;
        float val = acc[a][t][r] + bias;
        dst[(size_t)((b * H + hh) * NTOK + n) * HD + d] = (__bf16)val;
      }
    }
  }
}

// ---------------- kernel 3: per-(window,head) attention ----------------
// block = 128 threads (4 waves); one block per (b, head).
__global__ void __launch_bounds__(128) wa_attn_kernel(const float* __restrict__ mask,
                                                      char* __restrict__ ws) {
  constexpr float SCALE = 0.17677669529663687f;  // 32^-0.5
  const __bf16* Qg = (const __bf16*)(ws + OFF_Q);
  const __bf16* Kg = (const __bf16*)(ws + OFF_K);
  const __bf16* Vg = (const __bf16*)(ws + OFF_V);
  __bf16*       AO = (__bf16*)(ws + OFF_AO);
  const float*  ct = (const float*)(ws + OFF_COS);
  const float*  st = (const float*)(ws + OFF_SIN);

  __shared__ __bf16 sQ[NP * HD];   // [tok][d], padded tokens zeroed
  __shared__ __bf16 sK[NP * HD];   // [tok][d]
  __shared__ __bf16 sVt[HD * NP];  // [d][tok] (transposed for B-fragments)
  __shared__ float  sS[NP * NP];   // scores
  __shared__ __bf16 sP[NP * NP];   // softmax probs

  const int tid  = threadIdx.x;
  const int lane = tid & 31;
  const int wave = tid >> 5;
  const int h    = lane >> 4;
  const int bh   = blockIdx.x;
  const int b    = bh >> 4;
  const int head = bh & 15;
  const size_t base = (size_t)bh * NTOK * HD;

  // zero the padded LDS tiles
  {
    unsigned int* z0 = (unsigned int*)sQ;
    unsigned int* z1 = (unsigned int*)sK;
    unsigned int* z2 = (unsigned int*)sVt;
    for (int i = tid; i < NP * HD / 2; i += 128) { z0[i] = 0; z1[i] = 0; z2[i] = 0; }
    unsigned int* z3 = (unsigned int*)sP;
    for (int i = tid; i < NP * NP / 2; i += 128) z3[i] = 0;
  }
  __syncthreads();

  // load q,k with RoPE (+scale on q); v transposed
  for (int i = tid; i < NTOK * HD; i += 128) {
    int n = i >> 5, d = i & 31, j = d & 15;
    float cs = ct[n * 16 + j], sn = st[n * 16 + j];
    float qv = (float)Qg[base + i];
    float qp = (float)Qg[base + (i ^ 16)];      // partner element d^16
    float qr = (d < 16) ? -qp : qp;
    sQ[i] = (__bf16)((qv * cs + qr * sn) * SCALE);
    float kv = (float)Kg[base + i];
    float kp = (float)Kg[base + (i ^ 16)];
    float kr = (d < 16) ? -kp : kp;
    sK[i] = (__bf16)(kv * cs + kr * sn);
    sVt[d * NP + n] = Vg[base + i];
  }
  __syncthreads();

  // S = q . k^T  : wave owns 16 query rows x 64 key cols (K = hd = 32, one WMMA each)
  {
    int qrow = wave * 16 + (lane & 15);
    v16bf a = frag_a_bf16(sQ + qrow * HD, h);
#pragma unroll
    for (int t = 0; t < 4; ++t) {
      int krow = t * 16 + (lane & 15);
      v16bf bb = frag_b_bf16(sK + krow * HD, h);
      v8f s = wmma_bf16(a, bb, vzero8());
#pragma unroll
      for (int r = 0; r < 8; ++r)
        sS[(wave * 16 + r + 8 * h) * NP + t * 16 + (lane & 15)] = s[r];
    }
  }
  __syncthreads();

  // mask + softmax: one thread per valid query row
  if (tid < NTOK) {
    const int n = tid;
    const float* mrow = mask + ((size_t)(b & 63) * NTOK + n) * NTOK;
    float mx = -1e30f;
    for (int m = 0; m < NTOK; ++m) {
      float v = sS[n * NP + m] + mrow[m];
      sS[n * NP + m] = v;
      mx = fmaxf(mx, v);
    }
    float sum = 0.f;
    for (int m = 0; m < NTOK; ++m) {
      float e = __expf(sS[n * NP + m] - mx);
      sS[n * NP + m] = e;
      sum += e;
    }
    float inv = 1.f / sum;
    for (int m = 0; m < NTOK; ++m)
      sP[n * NP + m] = (__bf16)(sS[n * NP + m] * inv);
  }
  __syncthreads();

  // O = P . V : wave owns 16 query rows; 2 d-tiles x 2 K-steps of 32 keys
  {
    v8f o[2];
    o[0] = vzero8();
    o[1] = vzero8();
    int prow = wave * 16 + (lane & 15);
#pragma unroll
    for (int ks = 0; ks < 2; ++ks) {
      v16bf a = frag_a_bf16(sP + prow * NP + ks * 32, h);
#pragma unroll
      for (int t = 0; t < 2; ++t) {
        int dcol = t * 16 + (lane & 15);
        v16bf bb = frag_b_bf16(sVt + dcol * NP + ks * 32, h);
        o[t] = wmma_bf16(a, bb, o[t]);
      }
    }
#pragma unroll
    for (int t = 0; t < 2; ++t) {
      int dcol = t * 16 + (lane & 15);
#pragma unroll
      for (int r = 0; r < 8; ++r) {
        int n = wave * 16 + r + 8 * h;
        if (n < NTOK)
          AO[(size_t)(b * NTOK + n) * C + head * HD + dcol] = (__bf16)o[t][r];
      }
    }
  }
}

// ---------------- kernel 4: output projection GEMM (LDS double-buffered) --------
__global__ void __launch_bounds__(128) wa_proj_kernel(const float* __restrict__ bproj,
                                                      const char* __restrict__ ws,
                                                      float* __restrict__ out) {
  const __bf16* wp = (const __bf16*)(ws + OFF_WPROJ);
  const __bf16* AO = (const __bf16*)(ws + OFF_AO);

  __shared__ __bf16 sA[2][128 * LSTR];
  __shared__ __bf16 sB[2][64 * LSTR];

  const int tid  = threadIdx.x;
  const int lane = tid & 31;
  const int wave = tid >> 5;
  const int h    = lane >> 4;
  const int rowblock = blockIdx.y * 128;
  const int colbase  = blockIdx.x * 64;

  const __bf16* agp = AO + (size_t)(rowblock + tid) * C;
  const __bf16* bgp = wp + (size_t)(colbase + (tid >> 1)) * C + (tid & 1) * 16;
  const int sa_off = tid * LSTR;
  const int sb_off = (tid >> 1) * LSTR + (tid & 1) * 16;

  u32x4 af[4], bf2[2];
#pragma unroll
  for (int i = 0; i < 4; ++i) af[i] = *reinterpret_cast<const u32x4*>(agp + 8 * i);
#pragma unroll
  for (int j = 0; j < 2; ++j) bf2[j] = *reinterpret_cast<const u32x4*>(bgp + 8 * j);
#pragma unroll
  for (int i = 0; i < 4; ++i) *reinterpret_cast<u32x4*>(&sA[0][sa_off + 8 * i]) = af[i];
#pragma unroll
  for (int j = 0; j < 2; ++j) *reinterpret_cast<u32x4*>(&sB[0][sb_off + 8 * j]) = bf2[j];
  __syncthreads();

  v8f acc[2][4];
#pragma unroll
  for (int a = 0; a < 2; ++a)
#pragma unroll
    for (int t = 0; t < 4; ++t) acc[a][t] = vzero8();

  for (int s = 0; s < 16; ++s) {
    const int cur = s & 1, nxt = cur ^ 1;
    if (s < 15) {
      const int kk = (s + 1) * 32;
#pragma unroll
      for (int i = 0; i < 4; ++i) af[i] = *reinterpret_cast<const u32x4*>(agp + kk + 8 * i);
#pragma unroll
      for (int j = 0; j < 2; ++j) bf2[j] = *reinterpret_cast<const u32x4*>(bgp + kk + 8 * j);
    }
    const __bf16* sa = sA[cur];
    const __bf16* sb = sB[cur];
    v16bf a0 = frag_a_bf16(sa + (wave * 32 + (lane & 15)) * LSTR, h);
    v16bf a1 = frag_a_bf16(sa + (wave * 32 + 16 + (lane & 15)) * LSTR, h);
#pragma unroll
    for (int t = 0; t < 4; ++t) {
      v16bf bb = frag_b_bf16(sb + (t * 16 + (lane & 15)) * LSTR, h);
      acc[0][t] = wmma_bf16(a0, bb, acc[0][t]);
      acc[1][t] = wmma_bf16(a1, bb, acc[1][t]);
    }
    if (s < 15) {
#pragma unroll
      for (int i = 0; i < 4; ++i) *reinterpret_cast<u32x4*>(&sA[nxt][sa_off + 8 * i]) = af[i];
#pragma unroll
      for (int j = 0; j < 2; ++j) *reinterpret_cast<u32x4*>(&sB[nxt][sb_off + 8 * j]) = bf2[j];
    }
    __syncthreads();
  }

#pragma unroll
  for (int t = 0; t < 4; ++t) {
    int c = colbase + t * 16 + (lane & 15);
    float bias = bproj[c];
#pragma unroll
    for (int a = 0; a < 2; ++a) {
#pragma unroll
      for (int r = 0; r < 8; ++r) {
        int row = rowblock + wave * 32 + a * 16 + r + 8 * h;
        out[(size_t)row * C + c] = acc[a][t][r] + bias;
      }
    }
  }
}

// ---------------- host launcher ----------------
extern "C" void kernel_launch(void* const* d_in, const int* in_sizes, int n_in,
                              void* d_out, int out_size, void* d_ws, size_t ws_size,
                              hipStream_t stream) {
  (void)in_sizes; (void)n_in; (void)out_size; (void)ws_size;
  const float* x     = (const float*)d_in[0];
  const float* mask  = (const float*)d_in[1];
  const float* Wqkv  = (const float*)d_in[2];
  const float* bqkv  = (const float*)d_in[3];
  const float* Wproj = (const float*)d_in[4];
  const float* bproj = (const float*)d_in[5];
  float* out = (float*)d_out;
  char*  ws  = (char*)d_ws;

  wa_prep_kernel<<<dim3((C3 * C + 255) / 256), dim3(256), 0, stream>>>(Wqkv, Wproj, ws);
  wa_qkv_kernel<<<dim3(C3 / 64, MROWS / 128), dim3(128), 0, stream>>>(x, bqkv, ws);
  wa_attn_kernel<<<dim3(BW * H), dim3(128), 0, stream>>>(mask, ws);
  wa_proj_kernel<<<dim3(C / 64, MROWS / 128), dim3(128), 0, stream>>>(bproj, ws, out);
}
